// IEL_66872640799355
// MI455X (gfx1250) — compile-verified, matrix operand order checked by reading, not run on hardware
//
#include <hip/hip_runtime.h>
#include <math.h>

// ---------------- static problem config ----------------
#define HW_    12544         // 112*112
#define NB_    2
#define NTOK_  25088         // NB_*HW_
#define WIMG_  112

typedef __attribute__((ext_vector_type(16))) __bf16 v16bf;
typedef __attribute__((ext_vector_type(8)))  __bf16 v8bf;
typedef __attribute__((ext_vector_type(8)))  float  v8f;

#if defined(__has_builtin)
#  if __has_builtin(__builtin_amdgcn_global_load_async_to_lds_b128)
#    define HAVE_ASYNC 1
#  endif
#  if __has_builtin(__builtin_amdgcn_s_wait_asynccnt)
#    define HAVE_WAITASYNC 1
#  endif
#endif

#if HAVE_ASYNC
// builtin signature (from hipcc diagnostic): param is int __vector(4) * in the
// global (AS1) address space; LDS side is the AS3 analogue.
typedef int v4i_as __attribute__((vector_size(16)));
typedef __attribute__((address_space(1))) v4i_as* gas1_p;
typedef __attribute__((address_space(3))) v4i_as* las3_p;
#define GASYNC_B128(gsrc, ldst) \
  __builtin_amdgcn_global_load_async_to_lds_b128((gas1_p)(gsrc), (las3_p)(ldst), 0, 0)
#endif

static inline int cdiv_h(int a, int b) { return (a + b - 1) / b; }

// ---------------- helper kernels ----------------

// planar fp32 [B,64,H,W] -> channels-last bf16 [N,64]
__global__ void k_cvt_in(const float* __restrict__ x, __bf16* __restrict__ xnc) {
  int idx = blockIdx.x * 256 + threadIdx.x;
  if (idx >= NB_ * 64 * HW_) return;
  int hw = idx % HW_;
  int c  = (idx / HW_) % 64;
  int b  = idx / (HW_ * 64);
  xnc[((size_t)(b * HW_ + hw)) * 64 + c] = (__bf16)x[idx];
}

// fp32 [CoReal,CiReal] -> padded bf16 [Cop,Cip] with per-group scatter + scale
__global__ void k_pack_w(const float* __restrict__ w, __bf16* __restrict__ out,
                         int CoR, int CoP, int CiR, int CiP, int Cop, int Cip,
                         float scale) {
  int idx = blockIdx.x * 256 + threadIdx.x;
  if (idx >= Cop * Cip) return;
  int col = idx % Cip, r = idx / Cip;
  int gr = r / CoP, rr = r % CoP;
  int gc = col / CiP, cc = col % CiP;
  int CiReal = (Cip / CiP) * CiR;
  float v = 0.f;
  if (rr < CoR && cc < CiR)
    v = w[(size_t)(gr * CoR + rr) * CiReal + (gc * CiR + cc)] * scale;
  out[idx] = (__bf16)v;
}

// 16x16 mean pool of ctx [N,ldc] (bf16, Creal channels) -> ctxp [B,Creal,49] f32
__global__ void k_pool(const __bf16* __restrict__ ctx, int ldc, int Creal,
                       float* __restrict__ ctxp) {
  int idx = blockIdx.x * 256 + threadIdx.x;
  int tot = NB_ * Creal * 49;
  if (idx >= tot) return;
  int l = idx % 49;
  int c = (idx / 49) % Creal;
  int b = idx / (49 * Creal);
  int py = l / 7, px = l % 7;
  float s = 0.f;
  for (int i = 0; i < 16; ++i)
    for (int j = 0; j < 16; ++j) {
      int n = b * HW_ + (py * 16 + i) * WIMG_ + (px * 16 + j);
      s += (float)ctx[(size_t)n * ldc + c];
    }
  ctxp[idx] = s * (1.f / 256.f);
}

// key[b][g][l(64 pad)][k(Kp pad)] bf16 = wk[g*hd+k,:] . ctxp[b,:,l]
__global__ void k_keypack(const float* __restrict__ wk, const float* __restrict__ ctxp,
                          __bf16* __restrict__ key, int hd, int Kp, int Cctx) {
  int idx = blockIdx.x * 256 + threadIdx.x;
  int tot = NB_ * 2 * 64 * Kp;
  if (idx >= tot) return;
  int k = idx % Kp;
  int l = (idx / Kp) % 64;
  int g = (idx / (Kp * 64)) % 2;
  int b = idx / (Kp * 64 * 2);
  float v = 0.f;
  if (l < 49 && k < hd) {
    for (int c = 0; c < Cctx; ++c)
      v += wk[(size_t)(g * hd + k) * Cctx + c] * ctxp[((size_t)b * Cctx + c) * 49 + l];
  }
  key[idx] = (__bf16)v;
}

// project 49 ctx scores -> 18 taps, softmax 9+9. wmat f32 [BG*HW,64], attn f32 [BG*HW,18]
__global__ void k_attn(const float* __restrict__ wmat, const float* __restrict__ wp,
                       const float* __restrict__ wpb, float* __restrict__ attn) {
  int idx = blockIdx.x * 256 + threadIdx.x;
  if (idx >= NB_ * 2 * HW_) return;
  const float* wr = wmat + (size_t)idx * 64;
  float s[49];
#pragma unroll
  for (int j = 0; j < 49; ++j) s[j] = wr[j];
  float lg[18];
#pragma unroll
  for (int o = 0; o < 18; ++o) {
    float a = wpb[o];
#pragma unroll
    for (int j = 0; j < 49; ++j) a += s[j] * wp[o * 49 + j];
    lg[o] = a;
  }
  float* out = attn + (size_t)idx * 18;
#pragma unroll
  for (int grp = 0; grp < 2; ++grp) {
    float mx = -1e30f;
#pragma unroll
    for (int o = 0; o < 9; ++o) mx = fmaxf(mx, lg[grp * 9 + o]);
    float e[9], sum = 0.f;
#pragma unroll
    for (int o = 0; o < 9; ++o) { e[o] = __expf(lg[grp * 9 + o] - mx); sum += e[o]; }
    float inv = 1.f / sum;
#pragma unroll
    for (int o = 0; o < 9; ++o) out[grp * 9 + o] = e[o] * inv;
  }
}

// per-pixel per-channel 3x3 dynamic depthwise gather
__global__ void k_dyndw(const __bf16* __restrict__ src, int lds_, int halfpad_src,
                        __bf16* __restrict__ out, int ldo, int halfpad_out,
                        const float* __restrict__ attn, int C, int half, int hdmix) {
  int idx = blockIdx.x * 256 + threadIdx.x;
  if (idx >= NTOK_ * C) return;
  int c = idx % C;
  int n = idx / C;
  int b = n / HW_, hw = n % HW_;
  int h = hw / WIMG_, w = hw % WIMG_;
  int inhalf = (c < half) ? 0 : 1;
  int cl = c - inhalf * half;
  int g = cl / hdmix;
  int scol = inhalf ? (halfpad_src + cl) : cl;
  int ocol = inhalf ? (halfpad_out + cl) : cl;
  const float* a = attn + (((size_t)(b * 2 + g)) * HW_ + hw) * 18 + inhalf * 9;
  float s = 0.f;
#pragma unroll
  for (int t = 0; t < 9; ++t) {
    int dh = t / 3 - 1, dw = t % 3 - 1;
    int hh = h + dh, ww = w + dw;
    float xv = 0.f;
    if ((unsigned)hh < 112u && (unsigned)ww < 112u)
      xv = (float)src[((size_t)(b * HW_ + hh * WIMG_ + ww)) * lds_ + scol];
    s += xv * a[t];
  }
  out[(size_t)n * ldo + ocol] = (__bf16)s;
}

__global__ void k_zero_cols(__bf16* __restrict__ buf, int ld, int c0, int c1) {
  int w = c1 - c0;
  int idx = blockIdx.x * 256 + threadIdx.x;
  if (idx >= NTOK_ * w) return;
  int c = c0 + idx % w;
  int n = idx / w;
  buf[(size_t)n * ld + c] = (__bf16)0.f;
}

// ---------------- WMMA GEMM ----------------
// Y[N,Cop] = A[N,K](bf16, row stride lda) x Bw[Cop,K](bf16 packed, row major = B^T)
// Block tile 128M x 64N, 8 waves as 4(M) x 2(N); each wave owns 2x2 16x16 f32 tiles
// (4 WMMA per 32-K step). A and B double-buffered in LDS; staging uses CDNA5
// global_load_async_to_lds_b128 (ASYNCcnt) when available.
// EPI: 0 bf16, 1 f32, 2 bf16 tanh(acc)+Res, 3 f32 planar scatter, 4 bf16 (tanh(acc)+Res)*Mul
template <int EPI>
__global__ __launch_bounds__(256) void k_gemm(
    const __bf16* __restrict__ A, int lda, long astride_b, long astride_g,
    const __bf16* __restrict__ Bw, long bstride,
    void* __restrict__ Yv, int ldy, long ystride,
    const __bf16* __restrict__ Res, int ldres,
    const __bf16* __restrict__ Mul, int ldmul,
    int K, int Cop) {
  __shared__ __bf16 sA[2][128][40];
  __shared__ __bf16 sB[2][64][40];
  const int t  = threadIdx.x;
  const int wv = t >> 5;
  const int ln = t & 31;
  const int lm = ln & 15;
  const int hi = ln >> 4;
  const int wr = wv >> 1;     // wave row 0..3 (32 M each)
  const int wc = wv & 1;      // wave col 0..1 (32 N each)
  const int z  = blockIdx.z;
  const __bf16* Az = A + (size_t)(z >> 1) * astride_b + (size_t)(z & 1) * astride_g;
  const __bf16* Bz = Bw + (size_t)z * bstride;
  const int mbase = blockIdx.x * 128;
  const int nbb   = blockIdx.y * 64;

  const int srow = t >> 2;    // staging: 16B chunk (row, chunk) mapping
  const int sch  = t & 3;

  auto stage = [&](int buf, int k0) {
#if HAVE_ASYNC
    const __bf16* g0 = Az + (size_t)(mbase + srow) * lda + k0 + sch * 8;
    GASYNC_B128(g0, &sA[buf][srow][sch * 8]);
    const __bf16* g1 = Az + (size_t)(mbase + 64 + srow) * lda + k0 + sch * 8;
    GASYNC_B128(g1, &sA[buf][64 + srow][sch * 8]);
    if (nbb + srow < Cop) {
      const __bf16* gb = Bz + (size_t)(nbb + srow) * K + k0 + sch * 8;
      GASYNC_B128(gb, &sB[buf][srow][sch * 8]);
    }
#else
    uint4 ra0 = *(const uint4*)(Az + (size_t)(mbase + srow) * lda + k0 + sch * 8);
    uint4 ra1 = *(const uint4*)(Az + (size_t)(mbase + 64 + srow) * lda + k0 + sch * 8);
    uint4 rb  = make_uint4(0u, 0u, 0u, 0u);
    if (nbb + srow < Cop)
      rb = *(const uint4*)(Bz + (size_t)(nbb + srow) * K + k0 + sch * 8);
    *(uint4*)&sA[buf][srow][sch * 8] = ra0;
    *(uint4*)&sA[buf][64 + srow][sch * 8] = ra1;
    *(uint4*)&sB[buf][srow][sch * 8] = rb;
#endif
  };

  auto sync_stage = [&]() {
#if HAVE_ASYNC
#  if HAVE_WAITASYNC
    __builtin_amdgcn_s_wait_asynccnt(0);
#  else
    asm volatile("s_wait_asynccnt 0" ::: "memory");
#  endif
#endif
    __syncthreads();
  };

  v8f acc[2][2] = {};

  auto compute = [&](int cur) {
    v16bf af[2], bv[2];
#pragma unroll
    for (int s = 0; s < 2; ++s) {
      const int ar = wr * 32 + s * 16 + lm;
      v8bf a0 = *(const v8bf*)&sA[cur][ar][hi * 8];
      v8bf a1 = *(const v8bf*)&sA[cur][ar][16 + hi * 8];
      const int br = wc * 32 + s * 16 + lm;
      v8bf b0 = *(const v8bf*)&sB[cur][br][hi * 8];
      v8bf b1 = *(const v8bf*)&sB[cur][br][16 + hi * 8];
#pragma unroll
      for (int j = 0; j < 8; ++j) {
        af[s][j] = a0[j]; af[s][j + 8] = a1[j];
        bv[s][j] = b0[j]; bv[s][j + 8] = b1[j];
      }
    }
#pragma unroll
    for (int s = 0; s < 2; ++s)
#pragma unroll
      for (int c = 0; c < 2; ++c)
        acc[s][c] = __builtin_amdgcn_wmma_f32_16x16x32_bf16(
            false, af[s], false, bv[c], (short)0, acc[s][c], false, false);
  };

  const int nk = K >> 5;   // K is always a multiple of 32 (>= 64)
  stage(0, 0);
  sync_stage();
  for (int kt = 0; kt < nk - 1; ++kt) {
    const int cur = kt & 1;
    stage(cur ^ 1, (kt + 1) << 5);
    compute(cur);
    sync_stage();
  }
  compute((nk - 1) & 1);

#pragma unroll
  for (int s = 0; s < 2; ++s) {
#pragma unroll
    for (int c = 0; c < 2; ++c) {
      const int colbase = nbb + wc * 32 + c * 16;
      if (colbase >= Cop) continue;
      const int ocol = colbase + lm;
#pragma unroll
      for (int i = 0; i < 8; ++i) {
        const int orow = mbase + wr * 32 + s * 16 + i + hi * 8;  // C/D layout
        const float v = acc[s][c][i];
        if (EPI == 0) {
          ((__bf16*)Yv)[(size_t)z * ystride + (size_t)orow * ldy + ocol] = (__bf16)v;
        } else if (EPI == 1) {
          ((float*)Yv)[(size_t)z * ystride + (size_t)orow * ldy + ocol] = v;
        } else if (EPI == 2) {
          float r = (float)Res[(size_t)orow * ldres + ocol];
          ((__bf16*)Yv)[(size_t)orow * ldy + ocol] = (__bf16)(tanhf(v) + r);
        } else if (EPI == 4) {
          float r = (float)Res[(size_t)orow * ldres + ocol];
          float m = (float)Mul[(size_t)orow * ldmul + ocol];
          ((__bf16*)Yv)[(size_t)orow * ldy + ocol] = (__bf16)((tanhf(v) + r) * m);
        } else {  // EPI == 3: planar fp32 scatter [B,64,H,W]
          int b = orow / HW_, hw = orow % HW_;
          ((float*)Yv)[((size_t)b * 64 + ocol) * HW_ + hw] = v;
        }
      }
    }
  }
}

// ---------------- host side ----------------

static void run_gemm(int epi,
                     const __bf16* A, int lda, long asb, long asg,
                     const __bf16* Bw, long bst,
                     void* Y, int ldy, long yst,
                     const __bf16* Res, int ldres,
                     const __bf16* Mul, int ldmul,
                     int Nrows, int K, int Cop, int nz, hipStream_t s) {
  dim3 g(Nrows / 128, (Cop + 63) / 64, nz), b(256, 1, 1);
  switch (epi) {
    case 0: k_gemm<0><<<g, b, 0, s>>>(A, lda, asb, asg, Bw, bst, Y, ldy, yst, Res, ldres, Mul, ldmul, K, Cop); break;
    case 1: k_gemm<1><<<g, b, 0, s>>>(A, lda, asb, asg, Bw, bst, Y, ldy, yst, Res, ldres, Mul, ldmul, K, Cop); break;
    case 2: k_gemm<2><<<g, b, 0, s>>>(A, lda, asb, asg, Bw, bst, Y, ldy, yst, Res, ldres, Mul, ldmul, K, Cop); break;
    case 4: k_gemm<4><<<g, b, 0, s>>>(A, lda, asb, asg, Bw, bst, Y, ldy, yst, Res, ldres, Mul, ldmul, K, Cop); break;
    default: k_gemm<3><<<g, b, 0, s>>>(A, lda, asb, asg, Bw, bst, Y, ldy, yst, Res, ldres, Mul, ldmul, K, Cop); break;
  }
}

extern "C" void kernel_launch(void* const* d_in, const int* in_sizes, int n_in,
                              void* d_out, int out_size, void* d_ws, size_t ws_size,
                              hipStream_t stream) {
  (void)in_sizes; (void)n_in; (void)out_size; (void)ws_size;
  const float* x      = (const float*)d_in[0];
  const float* pin_w  = (const float*)d_in[1];
  const float* ctx_w  = (const float*)d_in[2];
  const float* ctx1_w = (const float*)d_in[3];
  const float* ctx2_w = (const float*)d_in[4];
  const float* pout_w = (const float*)d_in[5];
  const float* cm0_q  = (const float*)d_in[6];
  const float* cm0_k  = (const float*)d_in[7];
  const float* cm0_p  = (const float*)d_in[8];
  const float* cm0_pb = (const float*)d_in[9];
  const float* cm0_o  = (const float*)d_in[10];
  const float* cm1_q  = (const float*)d_in[11];
  const float* cm1_k  = (const float*)d_in[12];
  const float* cm1_p  = (const float*)d_in[13];
  const float* cm1_pb = (const float*)d_in[14];
  const float* cm1_o  = (const float*)d_in[15];
  const float* cm2_q  = (const float*)d_in[16];
  const float* cm2_k  = (const float*)d_in[17];
  const float* cm2_p  = (const float*)d_in[18];
  const float* cm2_pb = (const float*)d_in[19];
  const float* cm2_o  = (const float*)d_in[20];

  // bump allocator over workspace
  char* wsb = (char*)d_ws;
  size_t off = 0;
  auto alloc = [&](size_t bytes) -> void* {
    off = (off + 255) & ~(size_t)255;
    void* p = wsb + off;
    off += bytes;
    return p;
  };
  const size_t N = NTOK_;

  __bf16* Xnc   = (__bf16*)alloc(N * 64 * 2);
  __bf16* bufA  = (__bf16*)alloc(N * 384 * 2);  // x_in   [N,384] halves at 0/192
  __bf16* bufB  = (__bf16*)alloc(N * 384 * 2);  // y0; later y1|y2 then prod|..
  __bf16* bufC  = (__bf16*)alloc(N * 384 * 2);  // x_dw
  __bf16* ctxdw = (__bf16*)alloc(N * 96 * 2);
  __bf16* q0    = (__bf16*)alloc(N * 192 * 2);  // later x1p
  __bf16* q1    = (__bf16*)alloc(N * 128 * 2);
  __bf16* q2    = (__bf16*)alloc(N * 128 * 2);
  __bf16* c1b   = (__bf16*)alloc(N * 64 * 2);
  __bf16* c2b   = (__bf16*)alloc(N * 64 * 2);
  float*  wbuf  = (float*)alloc((size_t)4 * HW_ * 64 * 4);
  float*  abuf  = (float*)alloc((size_t)4 * HW_ * 18 * 4);
  float*  ctxp  = (float*)alloc((size_t)NB_ * 86 * 49 * 4);
  __bf16* key0  = (__bf16*)alloc((size_t)4 * 64 * 96 * 2);
  __bf16* key1  = (__bf16*)alloc((size_t)4 * 64 * 64 * 2);
  __bf16* key2  = (__bf16*)alloc((size_t)4 * 64 * 64 * 2);
  __bf16* WpinP = (__bf16*)alloc((size_t)384 * 64 * 2);
  __bf16* WctxP = (__bf16*)alloc((size_t)96 * 384 * 2);
  __bf16* W0qP  = (__bf16*)alloc((size_t)192 * 384 * 2);
  __bf16* W0oP  = (__bf16*)alloc((size_t)384 * 384 * 2);
  __bf16* Wc1P  = (__bf16*)alloc((size_t)64 * 192 * 2);
  __bf16* Wc2P  = (__bf16*)alloc((size_t)64 * 192 * 2);
  __bf16* W1qP  = (__bf16*)alloc((size_t)128 * 192 * 2);
  __bf16* W2qP  = (__bf16*)alloc((size_t)128 * 192 * 2);
  __bf16* W1oP  = (__bf16*)alloc((size_t)192 * 192 * 2);
  __bf16* W2oP  = (__bf16*)alloc((size_t)192 * 192 * 2);
  __bf16* WpoP  = (__bf16*)alloc((size_t)64 * 192 * 2);

  const float s0 = 1.0f / sqrtf(86.0f);
  const float s1 = 1.0f / sqrtf(43.0f);

  // ---- input conversion + weight packing ----
  k_cvt_in<<<cdiv_h(NB_ * 64 * HW_, 256), 256, 0, stream>>>(x, Xnc);
  auto pack = [&](const float* src, __bf16* dst, int CoR, int CoP, int CiR, int CiP,
                  int Cop, int Cip, float sc) {
    k_pack_w<<<cdiv_h(Cop * Cip, 256), 256, 0, stream>>>(src, dst, CoR, CoP, CiR, CiP, Cop, Cip, sc);
  };
  pack(pin_w,  WpinP, 172, 192,  64,  64, 384,  64, 1.f);
  pack(ctx_w,  WctxP,  86,  96, 172, 192,  96, 384, 1.f);
  pack(cm0_q,  W0qP,   86,  96, 172, 192, 192, 384, s0);
  pack(cm0_o,  W0oP,  172, 192, 172, 192, 384, 384, 1.f);
  pack(ctx1_w, Wc1P,   43,  64, 172, 192,  64, 192, 1.f);
  pack(ctx2_w, Wc2P,   43,  64, 172, 192,  64, 192, 1.f);
  pack(cm1_q,  W1qP,   43,  64, 172, 192, 128, 192, s1);
  pack(cm2_q,  W2qP,   43,  64, 172, 192, 128, 192, s1);
  pack(cm1_o,  W1oP,  172, 192, 172, 192, 192, 192, 1.f);
  pack(cm2_o,  W2oP,  172, 192, 172, 192, 192, 192, 1.f);
  pack(pout_w, WpoP,   64,  64, 172, 192,  64, 192, 1.f);

  // ---- stage 0 ----
  run_gemm(0, Xnc, 64, 0, 0, WpinP, 0, bufA, 384, 0, nullptr, 0, nullptr, 0,
           NTOK_, 64, 384, 1, stream);
  run_gemm(0, bufA, 384, 0, 0, WctxP, 0, ctxdw, 96, 0, nullptr, 0, nullptr, 0,
           NTOK_, 384, 96, 1, stream);
  k_pool<<<cdiv_h(NB_ * 86 * 49, 256), 256, 0, stream>>>(ctxdw, 96, 86, ctxp);
  k_keypack<<<cdiv_h(NB_ * 2 * 64 * 96, 256), 256, 0, stream>>>(cm0_k, ctxp, key0, 86, 96, 86);
  run_gemm(0, bufA, 384, 0, 0, W0qP, 0, q0, 192, 0, nullptr, 0, nullptr, 0,
           NTOK_, 384, 192, 1, stream);
  run_gemm(1, q0, 192, (long)HW_ * 192, 96, key0, (long)64 * 96,
           wbuf, 64, (long)HW_ * 64, nullptr, 0, nullptr, 0, HW_, 96, 64, 4, stream);
  k_attn<<<cdiv_h(NB_ * 2 * HW_, 256), 256, 0, stream>>>(wbuf, cm0_p, cm0_pb, abuf);
  k_dyndw<<<cdiv_h(NTOK_ * 344, 256), 256, 0, stream>>>(bufA, 384, 192, bufB, 384, 192,
                                                        abuf, 344, 172, 86);
  k_zero_cols<<<cdiv_h(NTOK_ * 20, 256), 256, 0, stream>>>(bufB, 384, 172, 192);
  k_zero_cols<<<cdiv_h(NTOK_ * 20, 256), 256, 0, stream>>>(bufB, 384, 364, 384);
  run_gemm(0, bufB, 384, 0, 0, W0oP, 0, bufC, 384, 0, nullptr, 0, nullptr, 0,
           NTOK_, 384, 384, 1, stream);

  // ---- branch 1 (x1 = bufC cols 0..191) ----
  run_gemm(0, bufC, 384, 0, 0, Wc1P, 0, c1b, 64, 0, nullptr, 0, nullptr, 0,
           NTOK_, 192, 64, 1, stream);
  k_pool<<<cdiv_h(NB_ * 43 * 49, 256), 256, 0, stream>>>(c1b, 64, 43, ctxp);
  k_keypack<<<cdiv_h(NB_ * 2 * 64 * 64, 256), 256, 0, stream>>>(cm1_k, ctxp, key1, 43, 64, 43);
  run_gemm(0, bufC, 384, 0, 0, W1qP, 0, q1, 128, 0, nullptr, 0, nullptr, 0,
           NTOK_, 192, 128, 1, stream);
  run_gemm(1, q1, 128, (long)HW_ * 128, 64, key1, (long)64 * 64,
           wbuf, 64, (long)HW_ * 64, nullptr, 0, nullptr, 0, HW_, 64, 64, 4, stream);
  k_attn<<<cdiv_h(NB_ * 2 * HW_, 256), 256, 0, stream>>>(wbuf, cm1_p, cm1_pb, abuf);
  __bf16* y1 = bufB;
  k_dyndw<<<cdiv_h(NTOK_ * 172, 256), 256, 0, stream>>>(bufC, 384, 86, y1, 192, 86,
                                                        abuf, 172, 86, 43);
  k_zero_cols<<<cdiv_h(NTOK_ * 20, 256), 256, 0, stream>>>(y1, 192, 172, 192);
  __bf16* x1p = q0;  // q0 dead
  run_gemm(2, y1, 192, 0, 0, W1oP, 0, x1p, 192, 0, bufC, 384, nullptr, 0,
           NTOK_, 192, 192, 1, stream);

  // ---- branch 2 (x2 = bufC cols 192..383) ----
  run_gemm(0, bufC + 192, 384, 0, 0, Wc2P, 0, c2b, 64, 0, nullptr, 0, nullptr, 0,
           NTOK_, 192, 64, 1, stream);
  k_pool<<<cdiv_h(NB_ * 43 * 49, 256), 256, 0, stream>>>(c2b, 64, 43, ctxp);
  k_keypack<<<cdiv_h(NB_ * 2 * 64 * 64, 256), 256, 0, stream>>>(cm2_k, ctxp, key2, 43, 64, 43);
  run_gemm(0, bufC + 192, 384, 0, 0, W2qP, 0, q2, 128, 0, nullptr, 0, nullptr, 0,
           NTOK_, 192, 128, 1, stream);
  run_gemm(1, q2, 128, (long)HW_ * 128, 64, key2, (long)64 * 64,
           wbuf, 64, (long)HW_ * 64, nullptr, 0, nullptr, 0, HW_, 64, 64, 4, stream);
  k_attn<<<cdiv_h(NB_ * 2 * HW_, 256), 256, 0, stream>>>(wbuf, cm2_p, cm2_pb, abuf);
  __bf16* y2 = bufB + (size_t)NTOK_ * 192;
  k_dyndw<<<cdiv_h(NTOK_ * 172, 256), 256, 0, stream>>>(bufC + 192, 384, 86, y2, 192, 86,
                                                        abuf, 172, 86, 43);
  k_zero_cols<<<cdiv_h(NTOK_ * 20, 256), 256, 0, stream>>>(y2, 192, 172, 192);

  // ---- fused: prod = (tanh(y2 x cm2_o^T) + x2) * x1p ----
  __bf16* prod = bufB;  // y1 region dead; A operand (y2) is disjoint
  run_gemm(4, y2, 192, 0, 0, W2oP, 0, prod, 192, 0, bufC + 192, 384, x1p, 192,
           NTOK_, 192, 192, 1, stream);

  // ---- output ----
  run_gemm(3, prod, 192, 0, 0, WpoP, 0, d_out, 0, 0, nullptr, 0, nullptr, 0,
           NTOK_, 192, 64, 1, stream);
}
